// SemStereo_60498909331982
// MI455X (gfx1250) — matrix operand order repaired and checked by reading, not compiled, not run on hardware
//
#include <hip/hip_runtime.h>
#include <hip/hip_bf16.h>
#include <math.h>

// ---- problem constants ----
#define G_     40
#define GP_    48           // group dim padded to 48 (multiple of 16)
#define CPG_   8
#define D_     48
#define H_     128
#define W_     240
#define MID_   16
#define HW_    (H_*W_)
#define KP     (27*GP_)     // 1296 padded K
#define KITER  41           // ceil(1296/32)
#define APITER 43           // KITER + 2 zero iters (pipeline + prefetch slack)

typedef __attribute__((ext_vector_type(16))) _Float16 v16h;
typedef __attribute__((ext_vector_type(8)))  _Float16 h8;
typedef __attribute__((ext_vector_type(8)))  float    v8f;

struct U32x2q { uint4 a, b; };   // 32B carrier for bit_cast -> v16h

// Exact parameter types of the async-to-LDS builtin (from the clang
// diagnostic): pointer to GCC-vector int4, AS(1) source / AS(3) dest.
typedef int v4i_t __attribute__((vector_size(16)));
typedef __attribute__((address_space(1))) v4i_t GV4;
typedef __attribute__((address_space(3))) v4i_t LV4;

__device__ __forceinline__ unsigned short f2h_bits(float x) {
    _Float16 h = (_Float16)x;
    return __builtin_bit_cast(unsigned short, h);
}
__device__ __forceinline__ float h2f(unsigned short u) {
    return (float)__builtin_bit_cast(_Float16, u);
}

// Async global->LDS 16B copy (ASYNCcnt-tracked).  AS pointers are formed by
// integer casts: global VA is identical in AS1; LDS offset = low 32 bits of
// the generic LDS address.
__device__ __forceinline__ void async_ld_b128(const uint4* gsrc, uint4* ldst) {
#if __has_builtin(__builtin_amdgcn_global_load_async_to_lds_b128)
    __builtin_amdgcn_global_load_async_to_lds_b128(
        (GV4*)(unsigned long long)gsrc,
        (LV4*)(unsigned)(unsigned long long)ldst, 0, 0);
#else
    unsigned lofs = (unsigned)(unsigned long long)ldst;
    asm volatile("global_load_async_to_lds_b128 %0, %1, off"
                 :: "v"(lofs), "v"((unsigned long long)gsrc) : "memory");
#endif
}
__device__ __forceinline__ void wait_async0() {
#if __has_builtin(__builtin_amdgcn_s_wait_asynccnt)
    __builtin_amdgcn_s_wait_asynccnt(0);
#else
    asm volatile("s_wait_asynccnt 0x0" ::: "memory");
#endif
}

// ------------------------------------------------------------------
// Kernel 0: pack conv1 weights [16][40][3][3][3] f32 into per-lane
// WMMA A-fragment order (16-bit A 16x32 layout), f16, for the padded
// K ordering k = tap*48 + g.  Zero for pad K (g>=40 or k>=KP) -- this
// also makes the tail WMMAs contribute exactly zero, so B needs no mask.
//   Apack flat index = ((iter*32)+lane)*16 + j
//   K(lane,j,iter)   = iter*32 + j + 16*(lane>>4)
// ------------------------------------------------------------------
__global__ void sem_pack_w(const float* __restrict__ w1,
                           unsigned short* __restrict__ apack) {
    int t = blockIdx.x * blockDim.x + threadIdx.x;
    if (t >= APITER * 32 * 16) return;
    int j    = t & 15;
    int lane = (t >> 4) & 31;
    int it   = t >> 9;
    int hk   = lane >> 4;
    int m    = lane & 15;
    int k    = it * 32 + j + 16 * hk;
    int tap  = k / GP_;
    int g    = k - tap * GP_;
    float v  = (k < KP && g < G_) ? w1[(m * G_ + g) * 27 + tap] : 0.0f;
    apack[t] = f2h_bits(v);
}

// ------------------------------------------------------------------
// Kernel 1: group-normalize both features -> f16, layout [h][w][g][c8]
// ------------------------------------------------------------------
__global__ void sem_normalize(const float* __restrict__ lf,
                              const float* __restrict__ rf,
                              unsigned short* __restrict__ f1n,
                              unsigned short* __restrict__ f2n) {
    int t = blockIdx.x * blockDim.x + threadIdx.x;
    const int N = G_ * HW_;
    if (t >= N) return;
    int g  = t / HW_;
    int hw = t - g * HW_;
    int base = g * CPG_ * HW_ + hw;

    float a[CPG_], b[CPG_];
    float sa = 0.f, sb = 0.f;
#pragma unroll
    for (int c = 0; c < CPG_; ++c) {
        a[c] = lf[base + c * HW_];
        b[c] = rf[base + c * HW_];
        sa += a[c] * a[c];
        sb += b[c] * b[c];
    }
    float ia = 1.0f / (sqrtf(sa) + 1e-5f);
    float ib = 1.0f / (sqrtf(sb) + 1e-5f);

    h8 pa, pb;
#pragma unroll
    for (int c = 0; c < CPG_; ++c) {
        pa[c] = (_Float16)(a[c] * ia);
        pb[c] = (_Float16)(b[c] * ib);
    }
    ((uint4*)f1n)[hw * G_ + g] = __builtin_bit_cast(uint4, pa);  // [h][w][g][c8]
    ((uint4*)f2n)[hw * G_ + g] = __builtin_bit_cast(uint4, pb);
}

// ------------------------------------------------------------------
// Kernel 2: shifted groupwise-correlation volume, f16 [d][h][w][g48]
// (g fastest, padded to 48, pad groups zeroed).
// One thread produces 8 g-values = one uint4; packed f16 math.
// ------------------------------------------------------------------
__global__ void sem_volume(const unsigned short* __restrict__ f1n,
                           const unsigned short* __restrict__ f2n,
                           uint4* __restrict__ volp) {
    int t = blockIdx.x * blockDim.x + threadIdx.x;
    const int N = D_ * HW_ * (GP_ / 8);
    if (t >= N) return;
    int q = t % (GP_ / 8);
    int r = t / (GP_ / 8);
    int w = r % W_;  r /= W_;
    int h = r % H_;
    int di = r / H_;

    uint4 out = make_uint4(0u, 0u, 0u, 0u);
    int dsp = di - 24;              // disparity in [-24, 24)
    int ws  = w - dsp;              // sample right feature at x - d
    if (q * 8 < G_ && ws >= 0 && ws < W_) {
        const uint4* f1q = (const uint4*)f1n;
        const uint4* f2q = (const uint4*)f2n;
        int b1 = (h * W_ + w)  * G_ + q * 8;
        int b2 = (h * W_ + ws) * G_ + q * 8;
        h8 o;
#pragma unroll
        for (int gg = 0; gg < 8; ++gg) {
            h8 x = __builtin_bit_cast(h8, f1q[b1 + gg]);
            h8 y = __builtin_bit_cast(h8, f2q[b2 + gg]);
            h8 p = x * y;                           // v_pk_mul_f16
            _Float16 s = ((p[0] + p[1]) + (p[2] + p[3])) +
                         ((p[4] + p[5]) + (p[6] + p[7]));
            o[gg] = s * (_Float16)0.125f;           // mean over 8 channels
        }
        out = __builtin_bit_cast(uint4, o);
    }
    volp[t] = out;
}

// ------------------------------------------------------------------
// Kernel 3: WMMA implicit-GEMM conv3d (40->16, 3x3x3) + fused LayerNorm.
// Block = 128 threads = 4 waves; each wave owns a 16-chan x 16-w tile at
// fixed (d,h).  LDS tile layout [dz*3+dy][x:68][g48] so each B fragment
// is 32 contiguous bytes -> two ds_load_b128 per WMMA.  The tile is
// staged with GLOBAL_LOAD_ASYNC_TO_LDS_B128 (ASYNCcnt); per-iteration
// fragment addresses come from a precomputed 84-entry LDS table; the
// next iteration's A/B loads are issued before the current WMMA.
// ------------------------------------------------------------------
#define TILE_X 68            // 64 block w + 2 halo + pad
#define TILE_Q (GP_ / 8)     // uint4 chunks per x position (6)
#define SLAB_C (TILE_X * TILE_Q)   // 408 uint4 chunks per (dz,dy) slab

__global__ __launch_bounds__(128)
void sem_conv1_ln(const uint4* __restrict__ volp,
                  const v16h* __restrict__ apk,
                  const float* __restrict__ lnw,
                  const float* __restrict__ lnb,
                  unsigned short* __restrict__ act) {
    __shared__ __align__(16) unsigned int sBu[9 * SLAB_C * 4];  // 58.75 KB
    __shared__ int sAddr[(KITER + 1) * 2];                      // 84 entries
    uint4* sB4 = (uint4*)sBu;

    const int d  = blockIdx.z;
    const int h  = blockIdx.y;
    const int wb = blockIdx.x * 64;
    const int tid = threadIdx.x;
    const uint4 z4 = make_uint4(0u, 0u, 0u, 0u);

    // ---- per-(it,hk) B-fragment address table (tail clamped) ----
    if (tid < (KITER + 1) * 2) {
        int itx = tid >> 1, hkk = tid & 1;
        int k0  = itx * 32 + 16 * hkk;
        if (k0 >= KP) k0 = KP - 16;          // clamp: A is zero there anyway
        int tap = k0 / GP_;
        int g0  = k0 - tap * GP_;            // 0, 16, 32
        int dz  = tap / 9;
        int rem = tap - dz * 9;
        int dy  = rem / 3;
        int dx  = rem - dy * 3;
        sAddr[tid] = ((dz * 3 + dy) * TILE_X + dx) * TILE_Q + (g0 >> 3);
    }

    // ---- async fill: 9 slabs, each contiguous in global & LDS ----
    for (int zy = 0; zy < 9; ++zy) {
        int zz = zy / 3, yy = zy - zz * 3;
        int dd = d + zz - 1;
        int hh = h + yy - 1;
        bool okdh = (dd >= 0 && dd < D_ && hh >= 0 && hh < H_);
        int lo = (wb == 0) ? TILE_Q : 0;            // gw >= 0
        int hi = (241 - wb) * TILE_Q;               // gw <= 239
        if (hi > SLAB_C) hi = SLAB_C;
        for (int i = tid; i < SLAB_C; i += 128) {
            if (okdh && i >= lo && i < hi)
                async_ld_b128(&volp[((dd * H_ + hh) * W_ + (wb - 1)) * TILE_Q + i],
                              &sB4[zy * SLAB_C + i]);
            else
                sB4[zy * SLAB_C + i] = z4;
        }
    }
    wait_async0();
    __syncthreads();

    const int lane = tid & 31;
    const int wave = tid >> 5;
    const int hk   = lane >> 4;    // K-half selector
    const int n    = lane & 15;    // N (spatial) index
    const int wloc = wave * 16;    // wave tile offset within block
    const int posq = (wloc + n) * TILE_Q;

    v8f c = {0.f, 0.f, 0.f, 0.f, 0.f, 0.f, 0.f, 0.f};

    // preload it = 0
    v16h a = apk[lane];
    int  bb = sAddr[hk] + posq;
    U32x2q u0{sB4[bb], sB4[bb + 1]};
    v16h b = __builtin_bit_cast(v16h, u0);

    for (int it = 0; it < KITER; ++it) {
        // issue next-iteration loads first (Apack/table padded past KITER)
        v16h an = apk[(it + 1) * 32 + lane];
        __builtin_prefetch((const void*)(apk + (it + 2) * 32 + lane), 0, 3);
        int bn = sAddr[2 * (it + 1) + hk] + posq;
        U32x2q un{sB4[bn], sB4[bn + 1]};

        c = __builtin_amdgcn_wmma_f32_16x16x32_f16(
                false, a, false, b, (short)0, c, false, false);

        a = an;
        b = __builtin_bit_cast(v16h, un);
    }

    // ---- fused LayerNorm over the 16 channels (= M dim of C tile) ----
    float s1 = 0.f, s2 = 0.f;
#pragma unroll
    for (int p = 0; p < 8; ++p) { s1 += c[p]; s2 += c[p] * c[p]; }
    s1 += __shfl_xor(s1, 16, 32);     // combine M=p and M=p+8 halves
    s2 += __shfl_xor(s2, 16, 32);
    float mean = s1 * (1.0f / 16.0f);
    float var  = s2 * (1.0f / 16.0f) - mean * mean;
    float rs   = rsqrtf(var + 1e-6f);

    int wg = wb + wloc + n;
    if (wg < W_) {
#pragma unroll
        for (int p = 0; p < 8; ++p) {
            int m = p + 8 * hk;
            float y = (c[p] - mean) * rs * lnw[m] + lnb[m];
            act[((m * D_ + d) * H_ + h) * W_ + wg] = f2h_bits(y);
        }
    }
}

// ------------------------------------------------------------------
// Kernel 4: conv3d 16->1, 3x3x3 (VALU; M=1 doesn't merit WMMA)
// ------------------------------------------------------------------
__global__ __launch_bounds__(256)
void sem_conv2(const unsigned short* __restrict__ act,
               const float* __restrict__ w2,
               float* __restrict__ cost) {
    __shared__ float sw[MID_ * 27];
    int tid = threadIdx.x;
    for (int i = tid; i < MID_ * 27; i += 256) sw[i] = w2[i];
    __syncthreads();

    int t = blockIdx.x * 256 + tid;
    if (t >= D_ * HW_) return;
    int w = t % W_;
    int r = t / W_;
    int h = r % H_;
    int d = r / H_;

    float acc = 0.f;
    for (int kz = 0; kz < 3; ++kz) {
        int dd = d + kz - 1; if (dd < 0 || dd >= D_) continue;
        for (int ky = 0; ky < 3; ++ky) {
            int hh = h + ky - 1; if (hh < 0 || hh >= H_) continue;
            for (int kx = 0; kx < 3; ++kx) {
                int ww = w + kx - 1; if (ww < 0 || ww >= W_) continue;
                int tap = (kz * 3 + ky) * 3 + kx;
                const unsigned short* ap = act + (dd * H_ + hh) * W_ + ww;
#pragma unroll
                for (int m = 0; m < MID_; ++m)
                    acc += h2f(ap[m * (D_ * HW_)]) * sw[m * 27 + tap];
            }
        }
    }
    cost[t] = acc;
}

// ------------------------------------------------------------------
// Kernel 5: top-2 softmax disparity regression
// ------------------------------------------------------------------
__global__ void sem_topk(const float* __restrict__ cost,
                         float* __restrict__ out) {
    int t = blockIdx.x * blockDim.x + threadIdx.x;
    if (t >= HW_) return;
    float v1 = -3.4e38f, v2 = -3.4e38f;
    int i1 = 0, i2 = 0;
    for (int di = 0; di < D_; ++di) {
        float v = cost[di * HW_ + t];
        if (v > v1)      { v2 = v1; i2 = i1; v1 = v; i1 = di; }
        else if (v > v2) { v2 = v;  i2 = di; }
    }
    float p1 = 1.0f / (1.0f + expf(v2 - v1));   // softmax over top-2
    float s1 = (float)(i1 - 24), s2 = (float)(i2 - 24);
    out[t] = s1 * p1 + s2 * (1.0f - p1);
}

// ------------------------------------------------------------------
// launch
// ------------------------------------------------------------------
extern "C" void kernel_launch(void* const* d_in, const int* in_sizes, int n_in,
                              void* d_out, int out_size, void* d_ws, size_t ws_size,
                              hipStream_t stream) {
    const float* left  = (const float*)d_in[0];
    const float* right = (const float*)d_in[1];
    const float* w1    = (const float*)d_in[2];
    const float* lnw   = (const float*)d_in[3];
    const float* lnb   = (const float*)d_in[4];
    const float* w2    = (const float*)d_in[5];
    (void)in_sizes; (void)n_in; (void)ws_size; (void)out_size;

    char* ws = (char*)d_ws;
    const size_t SZ_F   = (size_t)HW_ * G_ * CPG_ * 2;        // 19.66 MB each
    const size_t SZ_VOL = (size_t)D_ * HW_ * GP_ * 2;         // 141.6 MB
    const size_t SZ_AP  = (size_t)APITER * 32 * 16 * 2;       // 44 KB
    const size_t SZ_ACT = (size_t)MID_ * D_ * HW_ * 2;        // 47.2 MB
    size_t off = 0;
    auto take = [&](size_t sz) { size_t o = off; off += (sz + 255) & ~(size_t)255; return o; };
    unsigned short* f1n   = (unsigned short*)(ws + take(SZ_F));
    unsigned short* f2n   = (unsigned short*)(ws + take(SZ_F));
    uint4*          volp  = (uint4*)(ws + take(SZ_VOL));
    unsigned short* apack = (unsigned short*)(ws + take(SZ_AP));
    unsigned short* act   = (unsigned short*)(ws + take(SZ_ACT));
    float*          cost  = (float*)(ws + take((size_t)D_ * HW_ * 4));

    {   // weight pack
        int n = APITER * 32 * 16;
        sem_pack_w<<<(n + 255) / 256, 256, 0, stream>>>(w1, apack);
    }
    {   // normalize
        int n = G_ * HW_;
        sem_normalize<<<(n + 255) / 256, 256, 0, stream>>>(left, right, f1n, f2n);
    }
    {   // cost volume
        int n = D_ * HW_ * (GP_ / 8);
        sem_volume<<<(n + 255) / 256, 256, 0, stream>>>(f1n, f2n, volp);
    }
    {   // WMMA conv1 + LayerNorm
        dim3 grid((W_ + 63) / 64, H_, D_);
        sem_conv1_ln<<<grid, 128, 0, stream>>>(volp, (const v16h*)apack, lnw, lnb, act);
    }
    {   // conv2
        int n = D_ * HW_;
        sem_conv2<<<(n + 255) / 256, 256, 0, stream>>>(act, w2, cost);
    }
    {   // top-2 regression
        sem_topk<<<(HW_ + 255) / 256, 256, 0, stream>>>(cost, (float*)d_out);
    }
}
